// VisionMamba_3100966387778
// MI455X (gfx1250) — matrix operand order repaired and checked by reading, not compiled, not run on hardware
//
#include <hip/hip_runtime.h>
#include <hip/hip_bf16.h>

// ---------------------------------------------------------------------------
// Vision Mamba forward for MI455X (gfx1250, wave32).
// Projections use v_wmma_f32_16x16x32_bf16 with branch-free inner loops:
//   - M always a multiple of 32 (1568 rows) -> no M guards
//   - ragged N handled by clamping the weight row index (stores guarded)
//   - ragged K handled by zero-padding the A operand (dt path) + clamped B
// Each wave computes a 32x16 block (2 WMMAs / K-step, B fragment reused).
// Scan / norms / conv are fp32 VALU (sequential recurrence, tiny reductions).
// ---------------------------------------------------------------------------

typedef __bf16 bf16;
typedef __attribute__((ext_vector_type(16))) bf16  v16bf;
typedef __attribute__((ext_vector_type(8)))  float v8f;

#define BB 8
#define LL 196
#define EE 384
#define DI 768
#define SS 16
#define RR 24
#define DBL_W 56          // R + 2S
#define DT_PAD 32         // zero-padded K for the dt GEMM
#define ROWS (BB * LL)    // 1568 = 49 * 32
#define NC 1000

// ---------------- WMMA GEMM: C[m][n] = sum_k A[m][k] * W[n][k] ----------------
// A: bf16 (M x K, stride lda), K % 32 == 0, M % 32 == 0.
// W: fp32 weights (N x KB, stride ldb), KB <= K even; cols KB..K-1 of A are 0.
// One 32x16 output block per wave; 4 waves per block.
// EPI: 0 = store fp32; 1 = x_proj epilogue (fp32 dbl + zero-padded bf16 dt);
//      2 = softplus(x + bias[n]).

template<int EPI>
__global__ __launch_bounds__(128) void gemm_bf16_wmma(
    const bf16*  __restrict__ A,   int lda,
    const float* __restrict__ W,   int ldb,
    float*       __restrict__ C,   int ldc,
    bf16*        __restrict__ Cbf, int ldcb,
    const float* __restrict__ bias,
    int M, int N, int K, int KB)
{
  const int lane = threadIdx.x & 31;
  const int wave = threadIdx.x >> 5;
  const int Nt   = (N + 15) >> 4;
  const int Mg   = M >> 5;                       // 32-row groups
  const int tile = blockIdx.x * 4 + wave;        // wave-uniform
  if (tile >= Mg * Nt) return;                   // whole-wave exit, EXEC stays full
  const int m0 = (tile / Nt) * 32;
  const int n0 = (tile % Nt) * 16;

  const int half = lane >> 4;                    // 0: lanes 0-15, 1: lanes 16-31
  const int lh   = lane & 15;
  const int akb  = half * 8;                     // A K sub-block base
  const int bkb  = half * 16;                    // B K sub-block base
  const int bn   = n0 + lh;
  const int bnc  = (bn < N) ? bn : (N - 1);      // branchless clamp for ragged N

  const bf16*  Ar0 = A + (long)(m0 + lh) * lda + akb;        // rows m0..m0+15
  const bf16*  Ar1 = A + (long)(m0 + 16 + lh) * lda + akb;   // rows m0+16..m0+31
  const float* Wr  = W + (long)bnc * ldb;

  v8f acc0 = {0.f, 0.f, 0.f, 0.f, 0.f, 0.f, 0.f, 0.f};
  v8f acc1 = {0.f, 0.f, 0.f, 0.f, 0.f, 0.f, 0.f, 0.f};

  for (int k0 = 0; k0 < K; k0 += 32) {
    if (k0 + 32 < KB) __builtin_prefetch(Wr + k0 + 32, 0, 3);

    // ---- A fragments: two b128 loads each (16-byte aligned by construction)
    union { uint4 q[2]; v16bf v; } a0, a1;
    a0.q[0] = *(const uint4*)(Ar0 + k0);
    a0.q[1] = *(const uint4*)(Ar0 + k0 + 16);
    a1.q[0] = *(const uint4*)(Ar1 + k0);
    a1.q[1] = *(const uint4*)(Ar1 + k0 + 16);

    // ---- B fragment: 8 aligned float2 loads, packed to bf16 in-register.
    // kk clamped to KB-2: only happens when A's columns there are zero.
    union { unsigned u[8]; v16bf v; } b;
    #pragma unroll
    for (int j = 0; j < 8; ++j) {
      int kk = k0 + (bkb >> 1) + 2 * j + ((j >= 4) ? 8 : 0);
      kk = (kk <= KB - 2) ? kk : (KB - 2);
      const float2 f = *(const float2*)(Wr + kk);
      union { bf16 h[2]; unsigned u; } p;
      p.h[0] = (bf16)f.x; p.h[1] = (bf16)f.y;
      b.u[j] = p.u;
    }

    acc0 = __builtin_amdgcn_wmma_f32_16x16x32_bf16(
        false, a0.v, false, b.v, (short)0, acc0, false, false);
    acc1 = __builtin_amdgcn_wmma_f32_16x16x32_bf16(
        false, a1.v, false, b.v, (short)0, acc1, false, false);
  }

  // ---- store: VGPR r holds M = mbase + r (mbase includes 8*half), N = n0 + (lane&15) ----
  const int cn = n0 + lh;
  #pragma unroll
  for (int g = 0; g < 2; ++g) {
    const v8f& acc = g ? acc1 : acc0;
    const int mbase = m0 + g * 16 + half * 8;
    #pragma unroll
    for (int r = 0; r < 8; ++r) {
      const int cm = mbase + r;
      float v = acc[r];
      if (EPI == 0) {
        if (cn < N) C[(long)cm * ldc + cn] = v;
      } else if (EPI == 1) {
        // x_proj epilogue: fp32 dbl (dt|B|C, width 56) + bf16 dt operand
        // zero-padded to width 32 for the K=32 dt GEMM.
        if (cn < N) C[(long)cm * ldc + cn] = v;
        if (cn < DT_PAD) Cbf[(long)cm * ldcb + cn] = (bf16)((cn < RR) ? v : 0.f);
      } else { // EPI == 2: softplus(x + bias)
        if (cn < N) {
          float t = v + bias[cn];
          C[(long)cm * ldc + cn] = (t > 20.f) ? t : log1pf(__expf(t));
        }
      }
    }
  }
}

// ---------------- patch embed: conv16x16 stride16 + bias + pos ----------------

__global__ __launch_bounds__(EE) void patch_embed_kernel(
    const float* __restrict__ x, const float* __restrict__ pw,
    const float* __restrict__ pb, const float* __restrict__ pos,
    float* __restrict__ h)
{
  const int bl = blockIdx.x;            // b*196 + l
  const int b  = bl / LL, l = bl % LL;
  const int py = l / 14,  px = l % 14;
  __shared__ float patch[768];          // 3*16*16 pixels
  for (int i = threadIdx.x; i < 768; i += EE) {
    int c = i >> 8, rem = i & 255;
    int ph = rem >> 4, pq = rem & 15;
    patch[i] = x[(((long)b * 3 + c) * 224 + (py * 16 + ph)) * 224 + (px * 16 + pq)];
  }
  __syncthreads();
  const int e = threadIdx.x;
  float s = pb[e];
  const float* w = pw + (long)e * 768;
  #pragma unroll 8
  for (int i = 0; i < 768; ++i) s += patch[i] * w[i];
  h[(long)bl * EE + e] = s + pos[l * EE + e];
}

// ---------------- residual add + RMSNorm (fused, bf16 and/or fp32 out) -------

__global__ __launch_bounds__(128) void add_rmsnorm_kernel(
    const float* __restrict__ h, float* __restrict__ residual,
    const float* __restrict__ w,
    bf16* __restrict__ out_bf, float* __restrict__ out_f, int first)
{
  const int row = blockIdx.x;
  const int tid = threadIdx.x;
  __shared__ float red[128];
  const float* hr = h + (long)row * EE;
  float*       rr = residual + (long)row * EE;
  float v[3]; float ss = 0.f;
  #pragma unroll
  for (int i = 0; i < 3; ++i) {
    int e = tid + i * 128;
    float r = first ? hr[e] : (hr[e] + rr[e]);
    rr[e] = r; v[i] = r; ss += r * r;
  }
  red[tid] = ss; __syncthreads();
  for (int s = 64; s > 0; s >>= 1) {
    if (tid < s) red[tid] += red[tid + s];
    __syncthreads();
  }
  float scale = rsqrtf(red[0] * (1.f / EE) + 1e-5f);
  #pragma unroll
  for (int i = 0; i < 3; ++i) {
    int e = tid + i * 128;
    float o = v[i] * scale * w[e];
    if (out_bf) out_bf[(long)row * EE + e] = (bf16)o;
    if (out_f)  out_f[(long)row * EE + e]  = o;
  }
}

// ---------------- causal depthwise conv (DCONV=4) + SiLU ----------------

__global__ void conv_silu_kernel(const float* __restrict__ xz,
                                 const float* __restrict__ cw,
                                 const float* __restrict__ cb,
                                 float* __restrict__ xs_f, bf16* __restrict__ xs_bf)
{
  int idx = blockIdx.x * blockDim.x + threadIdx.x;
  if (idx >= BB * LL * DI) return;
  int c = idx % DI;
  int l = (idx / DI) % LL;
  int b = idx / (DI * LL);
  const float* base = xz + (long)b * LL * (2 * DI) + c;   // xs = xz[:, :DI]
  float acc = cb[c];
  #pragma unroll
  for (int j = 0; j < 4; ++j) {
    int ll = l - 3 + j;
    float xv = (ll >= 0) ? base[(long)ll * (2 * DI)] : 0.f;
    acc += xv * cw[c * 4 + j];
  }
  float s = acc / (1.f + __expf(-acc));                   // silu
  xs_f[idx]  = s;
  xs_bf[idx] = (bf16)s;
}

// ---------------- selective scan + D skip + SiLU(z) gate (-> bf16) ----------

__global__ __launch_bounds__(256) void scan_kernel(
    const float* __restrict__ xs, const float* __restrict__ dt,
    const float* __restrict__ dbl, const float* __restrict__ xz,
    const float* __restrict__ Alog, const float* __restrict__ Dp,
    bf16* __restrict__ yg)
{
  const int blk = blockIdx.x;                 // 3 blocks per batch element
  const int b   = blk / 3;
  const int c   = (blk % 3) * 256 + threadIdx.x;
  __shared__ float sB[SS], sC[SS];
  float Ac[SS], hst[SS];
  #pragma unroll
  for (int s = 0; s < SS; ++s) { Ac[s] = -__expf(Alog[c * SS + s]); hst[s] = 0.f; }
  const float Dc = Dp[c];
  for (int l = 0; l < LL; ++l) {
    long row = (long)b * LL + l;
    if (threadIdx.x < SS)            sB[threadIdx.x]      = dbl[row * DBL_W + RR + threadIdx.x];
    else if (threadIdx.x < 2 * SS)   sC[threadIdx.x - SS] = dbl[row * DBL_W + RR + SS + (threadIdx.x - SS)];
    __syncthreads();
    float x = xs[row * DI + c];
    float d = dt[row * DI + c];
    float y = 0.f;
    #pragma unroll
    for (int s = 0; s < SS; ++s) {
      float dA = __expf(d * Ac[s]);
      hst[s] = dA * hst[s] + (d * sB[s]) * x;
      y += hst[s] * sC[s];
    }
    y += x * Dc;
    float z = xz[row * (2 * DI) + DI + c];
    float g = z / (1.f + __expf(-z));
    yg[row * DI + c] = (bf16)(y * g);
    __syncthreads();
  }
}

// ---------------- mean pool over L ----------------

__global__ void pool_kernel(const float* __restrict__ hf, float* __restrict__ pooled)
{
  int i = blockIdx.x * blockDim.x + threadIdx.x;
  if (i >= BB * EE) return;
  int b = i / EE, e = i % EE;
  float s = 0.f;
  for (int l = 0; l < LL; ++l) s += hf[((long)b * LL + l) * EE + e];
  pooled[i] = s * (1.f / LL);
}

// ---------------- classification head ----------------

__global__ void head_kernel(const float* __restrict__ pooled,
                            const float* __restrict__ hw,
                            const float* __restrict__ hb,
                            float* __restrict__ out)
{
  int i = blockIdx.x * blockDim.x + threadIdx.x;
  if (i >= BB * NC) return;
  int b = i / NC, n = i % NC;
  float s = hb[n];
  const float* p = pooled + b * EE;
  const float* w = hw + (long)n * EE;
  #pragma unroll 4
  for (int k = 0; k < EE; ++k) s += p[k] * w[k];
  out[i] = s;
}

// ---------------------------------------------------------------------------

static inline size_t alignup(size_t x) { return (x + 255) & ~(size_t)255; }

extern "C" void kernel_launch(void* const* d_in, const int* in_sizes, int n_in,
                              void* d_out, int out_size, void* d_ws, size_t ws_size,
                              hipStream_t stream) {
  (void)in_sizes; (void)n_in; (void)out_size; (void)ws_size;

  const float* x_img    = (const float*)d_in[0];
  const float* patch_w  = (const float*)d_in[1];
  const float* patch_b  = (const float*)d_in[2];
  const float* pos_emb  = (const float*)d_in[3];
  const float* norm_ws  = (const float*)d_in[4];
  const float* in_proj  = (const float*)d_in[5];
  const float* conv_w   = (const float*)d_in[6];
  const float* conv_b   = (const float*)d_in[7];
  const float* x_proj   = (const float*)d_in[8];
  const float* dt_w     = (const float*)d_in[9];
  const float* dt_b     = (const float*)d_in[10];
  const float* A_log    = (const float*)d_in[11];
  const float* D_ssm    = (const float*)d_in[12];
  const float* out_proj = (const float*)d_in[13];
  const float* norm_f_w = (const float*)d_in[14];
  const float* head_w   = (const float*)d_in[15];
  const float* head_b   = (const float*)d_in[16];
  float* out = (float*)d_out;

  // ---- workspace layout (bump allocator) ----
  char* ws = (char*)d_ws;
  size_t off = 0;
  auto alloc = [&](size_t bytes) -> void* {
    void* p = ws + off; off = alignup(off + bytes); return p;
  };
  float* hbuf   = (float*)alloc((size_t)ROWS * EE * 4);        // layer output / input
  float* resid  = (float*)alloc((size_t)ROWS * EE * 4);        // fp32 residual stream
  bf16*  hn_bf  = (bf16*) alloc((size_t)ROWS * EE * 2);        // rmsnorm out (GEMM A)
  float* xz     = (float*)alloc((size_t)ROWS * 2 * DI * 4);    // in_proj out
  float* xs_f   = (float*)alloc((size_t)ROWS * DI * 4);        // conv+silu out
  bf16*  xs_bf  = (bf16*) alloc((size_t)ROWS * DI * 2);
  float* dbl    = (float*)alloc((size_t)ROWS * DBL_W * 4);     // x_proj out (dt|B|C)
  bf16*  dt_bf  = (bf16*) alloc((size_t)ROWS * DT_PAD * 2);    // dt operand, K-padded w/ zeros
  float* dtb    = (float*)alloc((size_t)ROWS * DI * 4);        // softplus(dt) out
  bf16*  yg_bf  = (bf16*) alloc((size_t)ROWS * DI * 2);        // gated scan out
  float* hf     = (float*)alloc((size_t)ROWS * EE * 4);        // final rmsnorm out
  float* pooled = (float*)alloc((size_t)BB * EE * 4);

  // ---- patch embed + pos ----
  patch_embed_kernel<<<ROWS, EE, 0, stream>>>(x_img, patch_w, patch_b, pos_emb, hbuf);

  // 32-row groups: Mg = 1568/32 = 49
  const int gIn  = (49 * 96 + 3) / 4;   // in_proj : N=1536 -> 96 tiles
  const int gXp  = (49 * 4  + 3) / 4;   // x_proj  : N=56   -> 4 tiles
  const int gDt  = (49 * 48 + 3) / 4;   // dt_proj : N=768  -> 48 tiles
  const int gOut = (49 * 24 + 3) / 4;   // out_proj: N=384  -> 24 tiles

  for (int i = 0; i < 24; ++i) {
    const float* nw  = norm_ws  + (long)i * EE;
    const float* wi  = in_proj  + (long)i * 2 * DI * EE;
    const float* cwi = conv_w   + (long)i * DI * 4;
    const float* cbi = conv_b   + (long)i * DI;
    const float* wx  = x_proj   + (long)i * DBL_W * DI;
    const float* wdt = dt_w     + (long)i * DI * RR;
    const float* bdt = dt_b     + (long)i * DI;
    const float* alg = A_log    + (long)i * DI * SS;
    const float* dss = D_ssm    + (long)i * DI;
    const float* wo  = out_proj + (long)i * EE * DI;

    // residual += h ; hn = rmsnorm(residual) -> bf16
    add_rmsnorm_kernel<<<ROWS, 128, 0, stream>>>(hbuf, resid, nw, hn_bf, nullptr, i == 0);

    // xz = hn @ wi^T   (1568 x 384 -> 1536)
    gemm_bf16_wmma<0><<<gIn, 128, 0, stream>>>(hn_bf, EE, wi, EE,
                                               xz, 2 * DI, nullptr, 0, nullptr,
                                               ROWS, 2 * DI, EE, EE);
    // xs = silu(dwconv_causal(xz[:, :DI]))
    conv_silu_kernel<<<(BB * LL * DI + 255) / 256, 256, 0, stream>>>(xz, cwi, cbi, xs_f, xs_bf);

    // dbl = xs @ wx^T  (1568 x 768 -> 56); also emits zero-padded bf16 dt operand
    gemm_bf16_wmma<1><<<gXp, 128, 0, stream>>>(xs_bf, DI, wx, DI,
                                               dbl, DBL_W, dt_bf, DT_PAD, nullptr,
                                               ROWS, DBL_W, DI, DI);
    // dt = softplus(dt_in @ wdt^T + bdt)  (K padded 24 -> 32; A cols 24..31 zero)
    gemm_bf16_wmma<2><<<gDt, 128, 0, stream>>>(dt_bf, DT_PAD, wdt, RR,
                                               dtb, DI, nullptr, 0, bdt,
                                               ROWS, DI, DT_PAD, RR);
    // selective scan + D skip + silu(z) gate -> bf16
    scan_kernel<<<BB * 3, 256, 0, stream>>>(xs_f, dtb, dbl, xz, alg, dss, yg_bf);

    // h = ygated @ wo^T  (1568 x 768 -> 384)
    gemm_bf16_wmma<0><<<gOut, 128, 0, stream>>>(yg_bf, DI, wo, DI,
                                                hbuf, EE, nullptr, 0, nullptr,
                                                ROWS, EE, DI, DI);
  }

  // final residual + rmsnorm -> hf (fp32)
  add_rmsnorm_kernel<<<ROWS, 128, 0, stream>>>(hbuf, resid, norm_f_w, nullptr, hf, 0);
  // mean pool over L
  pool_kernel<<<(BB * EE + 255) / 256, 256, 0, stream>>>(hf, pooled);
  // head
  head_kernel<<<(BB * NC + 255) / 256, 256, 0, stream>>>(pooled, head_w, head_b, out);
}
